// ClusterAttention_51015621542352
// MI455X (gfx1250) — matrix operand order repaired and checked by reading, not compiled, not run on hardware
//
#include <hip/hip_runtime.h>

typedef __attribute__((ext_vector_type(16))) _Float16 v16h;
typedef __attribute__((ext_vector_type(8)))  float    v8f;
typedef int v4i_b __attribute__((vector_size(16)));

constexpr int KC  = 256;   // clusters
constexpr int MP  = 256;   // points per cluster
constexpr int CD  = 512;   // feature dim
constexpr int NH  = 8;     // heads
constexpr int HD  = 64;    // head dim
constexpr int TCD = 1536;  // 3*CD

// ---------------- CDNA5 async global->LDS copy (guarded) ----------------

#if defined(__has_builtin)
#if __has_builtin(__builtin_amdgcn_global_load_async_to_lds_b128)
#define HAVE_ASYNC_LDS 1
#endif
#endif
#ifndef HAVE_ASYNC_LDS
#define HAVE_ASYNC_LDS 0
#endif

__device__ inline void copy_b128_to_lds(const _Float16* g, _Float16* l) {
#if HAVE_ASYNC_LDS
  v4i_b* gg = (v4i_b*)const_cast<_Float16*>(g);
  v4i_b* ll = (v4i_b*)l;
  __builtin_amdgcn_global_load_async_to_lds_b128(
      (__attribute__((address_space(1))) v4i_b*)gg,
      (__attribute__((address_space(3))) v4i_b*)ll, 0, 0);
#else
  *(uint4*)l = *(const uint4*)g;
#endif
}

__device__ inline void async_wait0() {
#if HAVE_ASYNC_LDS
#if __has_builtin(__builtin_amdgcn_s_wait_asynccnt)
  __builtin_amdgcn_s_wait_asynccnt(0);
#else
  asm volatile("s_wait_asynccnt 0" ::: "memory");
#endif
#endif
}

// ---------------- WMMA helpers ----------------

__device__ inline v8f v8f_zero() {
  v8f z;
#pragma unroll
  for (int i = 0; i < 8; ++i) z[i] = 0.0f;
  return z;
}

__device__ inline v8f wmma_f16(v16h a, v16h b, v8f c) {
  return __builtin_amdgcn_wmma_f32_16x16x32_f16(false, a, false, b, (short)0, c,
                                                false, false);
}

// A-matrix 16x32 f16 fragment: lane row = lane&15,
// lanes 0-15 hold K=0..15; lanes 16-31 hold K=16..31.
__device__ inline v16h load_afrag(const _Float16* rowchunk, int lane) {
  int off0 = (lane & 16) ? 16 : 0;
  union { v16h v; uint4 u[2]; } r;
  r.u[0] = *(const uint4*)(rowchunk + off0);
  r.u[1] = *(const uint4*)(rowchunk + off0 + 8);
  return r.v;
}

// B-matrix 32x16 f16 fragment: N = lane&15, lanes 0-15 hold K=0..15,
// lanes 16-31 hold K=16..31; B stored transposed (K innermost), 32B aligned.
__device__ inline v16h load_bfrag(const _Float16* colbase, int lane) {
  const _Float16* p = colbase + ((lane & 16) ? 16 : 0);
  return *(const v16h*)p;
}

// ---------------- pos preprocessing ----------------

__global__ void posmax_kernel(const float* __restrict__ pos, float* __restrict__ pm) {
  __shared__ float s0[256], s1[256];
  int t = threadIdx.x;
  float m0 = 0.f, m1 = 0.f;
  for (int i = t; i < KC * MP; i += 256) {
    m0 = fmaxf(m0, pos[i * 2 + 0]);
    m1 = fmaxf(m1, pos[i * 2 + 1]);
  }
  s0[t] = m0; s1[t] = m1; __syncthreads();
  for (int s = 128; s > 0; s >>= 1) {
    if (t < s) { s0[t] = fmaxf(s0[t], s0[t + s]); s1[t] = fmaxf(s1[t], s1[t + s]); }
    __syncthreads();
  }
  if (t == 0) { pm[0] = s0[0]; pm[1] = s1[0]; }
}

// trow[k,m,h] = pn . w_h                       (query-side, subtracted)
// tcol[k,m,h] = pn . w_h + b_h + mask_bias     (key-side, added)
__global__ void posproj_kernel(const float* __restrict__ pos,
                               const float* __restrict__ pos_w,
                               const float* __restrict__ pos_b,
                               const int* __restrict__ mask,
                               const float* __restrict__ pm,
                               float* __restrict__ trow, float* __restrict__ tcol) {
  int k = blockIdx.x, m = threadIdx.x;
  int idx = k * MP + m;
  float p0 = pos[idx * 2 + 0] / pm[0];
  float p1 = pos[idx * 2 + 1] / pm[1];
  float mb = mask[idx] ? 0.0f : -100.0f;
#pragma unroll
  for (int h = 0; h < NH; ++h) {
    float t = p0 * pos_w[h] + p1 * pos_w[NH + h];
    trow[idx * NH + h] = t;
    tcol[idx * NH + h] = t + pos_b[h] + mb;
  }
}

__global__ void cvt_kernel(const float* __restrict__ src, _Float16* __restrict__ dst, int n) {
  int i = blockIdx.x * 256 + threadIdx.x;
  if (i < n) dst[i] = (_Float16)src[i];
}

// ---------------- GEMM1: feat(65536x512) @ qkv_w(512x1536) -> q,k,v f16 ----------------
// grid (12, 512), 256 threads (8 waves, 2x4 wave grid over a 128x128 tile).
// Double-buffered LDS, register-staged loads, 1 barrier / K-step.

__global__ void gemm1_kernel(const float* __restrict__ feat,
                             const _Float16* __restrict__ w16,
                             const float* __restrict__ qkv_b,
                             _Float16* __restrict__ qf,
                             _Float16* __restrict__ kf,
                             _Float16* __restrict__ vf) {
  __shared__ __align__(32) _Float16 smA[2][128 * 32];   // [row][k]
  __shared__ __align__(32) _Float16 smBT[2][128 * 32];  // [n][k] (transposed)
  int col0 = blockIdx.x * 128;
  int row0 = blockIdx.y * 128;
  int tid = threadIdx.x, lane = tid & 31, wid = tid >> 5;
  int wm = wid >> 2, wn = wid & 3;
  int ar = tid >> 3, ac = (tid & 7) * 4;   // A staging map (4 passes, r+=32)
  int br = tid >> 4, bc = (tid & 15) * 8;  // B staging map (2 passes, r+=16)

  v8f acc[4][2];
#pragma unroll
  for (int mt = 0; mt < 4; ++mt)
#pragma unroll
    for (int nt = 0; nt < 2; ++nt) acc[mt][nt] = v8f_zero();

  float4 fA[4];
  uint4 uB[2];
  auto load_tile = [&](int ki) {
    int kt = ki * 32;
#pragma unroll
    for (int p = 0; p < 4; ++p)
      fA[p] = *(const float4*)(feat + (size_t)(row0 + ar + p * 32) * CD + kt + ac);
#pragma unroll
    for (int p = 0; p < 2; ++p)
      uB[p] = *(const uint4*)(w16 + (size_t)(kt + br + p * 16) * TCD + col0 + bc);
  };
  auto store_tile = [&](int buf) {
#pragma unroll
    for (int p = 0; p < 4; ++p) {
      _Float16* d = &smA[buf][(ar + p * 32) * 32 + ac];
      d[0] = (_Float16)fA[p].x; d[1] = (_Float16)fA[p].y;
      d[2] = (_Float16)fA[p].z; d[3] = (_Float16)fA[p].w;
    }
#pragma unroll
    for (int p = 0; p < 2; ++p) {
      union { uint4 u; _Float16 h[8]; } t; t.u = uB[p];
#pragma unroll
      for (int j = 0; j < 8; ++j) smBT[buf][(bc + j) * 32 + br + p * 16] = t.h[j];
    }
  };

  load_tile(0);
  store_tile(0);
  __syncthreads();

  for (int it = 0; it < 16; ++it) {
    int cur = it & 1;
    bool more = (it + 1) < 16;
    if (more) load_tile(it + 1);  // overlap next-tile global loads with WMMA
    v16h bfr[2];
#pragma unroll
    for (int nt = 0; nt < 2; ++nt) {
      int n = wn * 32 + nt * 16 + (lane & 15);
      bfr[nt] = load_bfrag(&smBT[cur][n * 32], lane);
    }
#pragma unroll
    for (int mt = 0; mt < 4; ++mt) {
      int r = wm * 64 + mt * 16 + (lane & 15);
      v16h af = load_afrag(&smA[cur][r * 32], lane);
#pragma unroll
      for (int nt = 0; nt < 2; ++nt) acc[mt][nt] = wmma_f16(af, bfr[nt], acc[mt][nt]);
    }
    if (more) {
      store_tile(cur ^ 1);
      __syncthreads();
    }
  }

  // block-uniform q/k/v selection (col0 lies fully inside one 512-col segment)
  int which = col0 >> 9;
  _Float16* dst = (which == 0) ? qf : (which == 1) ? kf : vf;
  float scale = (which == 0) ? 0.125f : 1.0f;  // hd^-0.5 folded into q
  int hi = (lane & 16) ? 8 : 0;
#pragma unroll
  for (int mt = 0; mt < 4; ++mt)
#pragma unroll
    for (int nt = 0; nt < 2; ++nt)
#pragma unroll
      for (int i = 0; i < 8; ++i) {
        int gr = row0 + wm * 64 + mt * 16 + i + hi;
        int gc = col0 + wn * 32 + nt * 16 + (lane & 15);
        int rem = gc & 511, h = rem >> 6, e = rem & 63;
        int kcl = gr >> 8, mm = gr & 255;
        size_t di = (((size_t)(kcl * NH + h)) * MP + mm) * HD + e;
        dst[di] = (_Float16)((acc[mt][nt][i] + qkv_b[gc]) * scale);
      }
}

// ---------------- Flash attention: one block per (cluster, head) ----------------

__global__ void attn_kernel(const _Float16* __restrict__ qf,
                            const _Float16* __restrict__ kf,
                            const _Float16* __restrict__ vf,
                            const float* __restrict__ trow,
                            const float* __restrict__ tcol,
                            _Float16* __restrict__ of16) {
  extern __shared__ char smem[];
  _Float16* smK  = (_Float16*)smem;            // 256*64
  _Float16* smVt = smK + MP * HD;              // 64*256
  _Float16* smP  = smVt + HD * MP;             // 8*32*32
  float* smCol = (float*)(smP + 8 * 32 * 32);  // 256
  float* smRow = smCol + MP;                   // 256

  int bx = blockIdx.x;
  int kcl = bx >> 3, h = bx & 7;
  int tid = threadIdx.x, lane = tid & 31, wid = tid >> 5;
  size_t base = ((size_t)(kcl * NH + h)) * MP * HD;
  const _Float16* Kg = kf + base;
  const _Float16* Vg = vf + base;
  const _Float16* Qg = qf + base;

  __builtin_prefetch(Qg + (size_t)tid * HD, 0, 0);  // global_prefetch_b8 for Q rows

#pragma unroll
  for (int i = 0; i < 8; ++i) {  // K: 2048 x b128, async copy into LDS
    int u = tid + i * 256, r = u >> 3, c = (u & 7) * 8;
    copy_b128_to_lds(Kg + r * HD + c, smK + r * HD + c);
  }
  {  // V transposed (needs lane transpose -> manual)
    int r = tid;
#pragma unroll
    for (int c8 = 0; c8 < 8; ++c8) {
      union { uint4 u; _Float16 x[8]; } t;
      t.u = *(const uint4*)(Vg + r * HD + c8 * 8);
#pragma unroll
      for (int j = 0; j < 8; ++j) smVt[(c8 * 8 + j) * MP + r] = t.x[j];
    }
  }
  smCol[tid] = tcol[((size_t)kcl * MP + tid) * NH + h];
  smRow[tid] = trow[((size_t)kcl * MP + tid) * NH + h];
  async_wait0();
  __syncthreads();

  int qrow0 = wid * 32;
  v16h qa[2][2];
#pragma unroll
  for (int mt = 0; mt < 2; ++mt)
#pragma unroll
    for (int kc = 0; kc < 2; ++kc)
      qa[mt][kc] = load_afrag(Qg + (size_t)(qrow0 + mt * 16 + (lane & 15)) * HD + kc * 32, lane);

  v8f oacc[2][4];
  float rm[2][8], rl[2][8];
#pragma unroll
  for (int mt = 0; mt < 2; ++mt) {
#pragma unroll
    for (int nt = 0; nt < 4; ++nt) oacc[mt][nt] = v8f_zero();
#pragma unroll
    for (int i = 0; i < 8; ++i) { rm[mt][i] = -1e30f; rl[mt][i] = 0.0f; }
  }
  _Float16* smPw = smP + wid * 32 * 32;
  int hi = (lane & 16) ? 8 : 0;

  for (int nb = 0; nb < 8; ++nb) {  // stream 32-key blocks
    v8f s[2][2];
#pragma unroll
    for (int mt = 0; mt < 2; ++mt)
#pragma unroll
      for (int nt = 0; nt < 2; ++nt) s[mt][nt] = v8f_zero();
#pragma unroll
    for (int nt = 0; nt < 2; ++nt) {
      int n = nb * 32 + nt * 16 + (lane & 15);
      v16h b0 = load_bfrag(smK + n * HD + 0, lane);
      v16h b1 = load_bfrag(smK + n * HD + 32, lane);
#pragma unroll
      for (int mt = 0; mt < 2; ++mt) {
        s[mt][nt] = wmma_f16(qa[mt][0], b0, s[mt][nt]);
        s[mt][nt] = wmma_f16(qa[mt][1], b1, s[mt][nt]);
      }
    }
    float c0 = smCol[nb * 32 + (lane & 15)];
    float c1 = smCol[nb * 32 + 16 + (lane & 15)];
#pragma unroll
    for (int mt = 0; mt < 2; ++mt)
#pragma unroll
      for (int i = 0; i < 8; ++i) {
        int arow = qrow0 + mt * 16 + i + hi;
        float rv = smRow[arow];
        float v0 = s[mt][0][i] + c0 - rv;
        float v1 = s[mt][1][i] + c1 - rv;
        float mx = fmaxf(v0, v1);
        mx = fmaxf(mx, __shfl_xor(mx, 1, 32));
        mx = fmaxf(mx, __shfl_xor(mx, 2, 32));
        mx = fmaxf(mx, __shfl_xor(mx, 4, 32));
        mx = fmaxf(mx, __shfl_xor(mx, 8, 32));
        float mnew = fmaxf(rm[mt][i], mx);
        float corr = __expf(rm[mt][i] - mnew);
        float p0 = __expf(v0 - mnew), p1 = __expf(v1 - mnew);
        float ps = p0 + p1;
        ps += __shfl_xor(ps, 1, 32);
        ps += __shfl_xor(ps, 2, 32);
        ps += __shfl_xor(ps, 4, 32);
        ps += __shfl_xor(ps, 8, 32);
        rl[mt][i] = rl[mt][i] * corr + ps;
        rm[mt][i] = mnew;
#pragma unroll
        for (int nt2 = 0; nt2 < 4; ++nt2) oacc[mt][nt2][i] *= corr;
        int prl = mt * 16 + i + hi;
        smPw[prl * 32 + (lane & 15)]      = (_Float16)p0;
        smPw[prl * 32 + 16 + (lane & 15)] = (_Float16)p1;
      }
    asm volatile("s_wait_dscnt 0" ::: "memory");  // wave-local P RAW through LDS
#pragma unroll
    for (int mt = 0; mt < 2; ++mt) {
      v16h pa = load_afrag(smPw + (mt * 16 + (lane & 15)) * 32, lane);
#pragma unroll
      for (int ntO = 0; ntO < 4; ++ntO) {
        v16h bv = load_bfrag(smVt + (ntO * 16 + (lane & 15)) * MP + nb * 32, lane);
        oacc[mt][ntO] = wmma_f16(pa, bv, oacc[mt][ntO]);
      }
    }
  }

#pragma unroll
  for (int mt = 0; mt < 2; ++mt)
#pragma unroll
    for (int i = 0; i < 8; ++i) {
      float inv = 1.0f / rl[mt][i];
      int grow = kcl * MP + qrow0 + mt * 16 + i + hi;
#pragma unroll
      for (int ntO = 0; ntO < 4; ++ntO) {
        int gcol = h * HD + ntO * 16 + (lane & 15);
        of16[(size_t)grow * CD + gcol] = (_Float16)(oacc[mt][ntO][i] * inv);
      }
    }
}

// ---------------- GEMM2: O(65536x512) @ proj_w(512x512) -> f32 out ----------------
// Double-buffered; A tile copied with async global->LDS (f16 straight copy).

__global__ void gemm2_kernel(const _Float16* __restrict__ of16,
                             const _Float16* __restrict__ wp16,
                             const float* __restrict__ proj_b,
                             float* __restrict__ out) {
  __shared__ __align__(32) _Float16 smA[2][128 * 32];
  __shared__ __align__(32) _Float16 smBT[2][128 * 32];
  int col0 = blockIdx.x * 128;
  int row0 = blockIdx.y * 128;
  int tid = threadIdx.x, lane = tid & 31, wid = tid >> 5;
  int wm = wid >> 2, wn = wid & 3;
  int br = tid >> 4, bc = (tid & 15) * 8;

  v8f acc[4][2];
#pragma unroll
  for (int mt = 0; mt < 4; ++mt)
#pragma unroll
    for (int nt = 0; nt < 2; ++nt) acc[mt][nt] = v8f_zero();

  uint4 uB[2];
  auto loadA_async = [&](int ki, int buf) {
    int kt = ki * 32;
#pragma unroll
    for (int p = 0; p < 2; ++p) {
      int u = tid + p * 256, r = u >> 2, c = (u & 3) * 8;
      copy_b128_to_lds(of16 + (size_t)(row0 + r) * CD + kt + c, &smA[buf][r * 32 + c]);
    }
  };
  auto loadB = [&](int ki) {
    int kt = ki * 32;
#pragma unroll
    for (int p = 0; p < 2; ++p)
      uB[p] = *(const uint4*)(wp16 + (size_t)(kt + br + p * 16) * CD + col0 + bc);
  };
  auto storeB = [&](int buf) {
#pragma unroll
    for (int p = 0; p < 2; ++p) {
      union { uint4 u; _Float16 h[8]; } t; t.u = uB[p];
#pragma unroll
      for (int j = 0; j < 8; ++j) smBT[buf][(bc + j) * 32 + br + p * 16] = t.h[j];
    }
  };

  loadA_async(0, 0);
  loadB(0);
  storeB(0);
  async_wait0();
  __syncthreads();

  for (int it = 0; it < 16; ++it) {
    int cur = it & 1;
    bool more = (it + 1) < 16;
    if (more) { loadA_async(it + 1, cur ^ 1); loadB(it + 1); }
    v16h bfr[2];
#pragma unroll
    for (int nt = 0; nt < 2; ++nt) {
      int n = wn * 32 + nt * 16 + (lane & 15);
      bfr[nt] = load_bfrag(&smBT[cur][n * 32], lane);
    }
#pragma unroll
    for (int mt = 0; mt < 4; ++mt) {
      int r = wm * 64 + mt * 16 + (lane & 15);
      v16h af = load_afrag(&smA[cur][r * 32], lane);
#pragma unroll
      for (int nt = 0; nt < 2; ++nt) acc[mt][nt] = wmma_f16(af, bfr[nt], acc[mt][nt]);
    }
    if (more) {
      storeB(cur ^ 1);
      async_wait0();
      __syncthreads();
    }
  }

  int hi = (lane & 16) ? 8 : 0;
#pragma unroll
  for (int mt = 0; mt < 4; ++mt)
#pragma unroll
    for (int nt = 0; nt < 2; ++nt)
#pragma unroll
      for (int i = 0; i < 8; ++i) {
        int gr = row0 + wm * 64 + mt * 16 + i + hi;
        int gc = col0 + wn * 32 + nt * 16 + (lane & 15);
        out[(size_t)gr * CD + gc] = acc[mt][nt][i] + proj_b[gc];
      }
}

// ---------------- launch ----------------

extern "C" void kernel_launch(void* const* d_in, const int* in_sizes, int n_in,
                              void* d_out, int out_size, void* d_ws, size_t ws_size,
                              hipStream_t stream) {
  const float* pos    = (const float*)d_in[0];
  const float* feat   = (const float*)d_in[1];
  const float* qkv_w  = (const float*)d_in[2];
  const float* qkv_b  = (const float*)d_in[3];
  const float* pos_w  = (const float*)d_in[4];
  const float* pos_b  = (const float*)d_in[5];
  const float* proj_w = (const float*)d_in[6];
  const float* proj_b = (const float*)d_in[7];
  const int*   mask   = (const int*)d_in[8];
  float* out = (float*)d_out;

  char* ws = (char*)d_ws;
  size_t off = 0;
  auto carve = [&](size_t bytes) -> char* {
    char* p = ws + off;
    off += (bytes + 255) & ~(size_t)255;
    return p;
  };
  const size_t qkv_bytes = (size_t)2 * KC * NH * MP * HD;  // 64 MB each
  _Float16* qf   = (_Float16*)carve(qkv_bytes);
  _Float16* kf   = (_Float16*)carve(qkv_bytes);
  _Float16* vf   = (_Float16*)carve(qkv_bytes);
  _Float16* of   = (_Float16*)carve((size_t)2 * KC * MP * CD);
  _Float16* wq16 = (_Float16*)carve((size_t)2 * CD * TCD);
  _Float16* wp16 = (_Float16*)carve((size_t)2 * CD * CD);
  float* trow = (float*)carve((size_t)4 * KC * MP * NH);
  float* tcol = (float*)carve((size_t)4 * KC * MP * NH);
  float* pm   = (float*)carve(256);
  (void)in_sizes; (void)n_in; (void)out_size; (void)ws_size;

  posmax_kernel<<<1, 256, 0, stream>>>(pos, pm);
  posproj_kernel<<<KC, MP, 0, stream>>>(pos, pos_w, pos_b, mask, pm, trow, tcol);
  cvt_kernel<<<(CD * TCD + 255) / 256, 256, 0, stream>>>(qkv_w, wq16, CD * TCD);
  cvt_kernel<<<(CD * CD + 255) / 256, 256, 0, stream>>>(proj_w, wp16, CD * CD);

  gemm1_kernel<<<dim3(TCD / 128, (KC * MP) / 128), 256, 0, stream>>>(
      feat, wq16, qkv_b, qf, kf, vf);

  const int attn_lds = (MP * HD + HD * MP + 8 * 32 * 32) * 2 + 2 * MP * 4;  // 83968 B
  attn_kernel<<<KC * NH, 256, attn_lds, stream>>>(qf, kf, vf, trow, tcol, of);

  gemm2_kernel<<<dim3(CD / 128, (KC * MP) / 128), 256, 0, stream>>>(
      of, wp16, proj_b, out);
}
